// quant_Linear_76673756168370
// MI455X (gfx1250) — matrix operand order repaired and checked by reading, not compiled, not run on hardware
//
#include <hip/hip_runtime.h>
#include <hip/hip_bf16.h>

// ---------------------------------------------------------------------------
// quantized linear: out[M,N] = x[M,K] * codebook[idx[N,K]]^T + bias[N]
//   M = B*S = 8192, K = DIN = 4096, N = DOUT = 4096
// Strategy: pre-swizzle x -> bf16 A-fragments and dequantized W -> bf16
// B-fragments in d_ws, then run a double-buffered LDS bf16 WMMA GEMM.
// ---------------------------------------------------------------------------

#define DIN   4096
#define DOUT  4096
#define MROWS 8192
#define KB    (DIN / 32)   // 128 k-blocks of depth 32

typedef __attribute__((ext_vector_type(16))) __bf16 v16bf;
typedef __attribute__((ext_vector_type(8)))  float  v8f;

#if defined(__gfx1250__) && __has_builtin(__builtin_amdgcn_global_load_async_to_lds_b128) && __has_builtin(__builtin_amdgcn_s_wait_asynccnt)
#define USE_ASYNC 1
// exact parameter type per hipcc diagnostic: int __attribute__((vector_size(16))) *
typedef int async_v4i __attribute__((vector_size(16)));
#else
#define USE_ASYNC 0
#endif

// ---------------------------------------------------------------------------
// Stage 1a: x (f32, row-major [M][K]) -> A_sw (bf16) in WMMA A-fragment order.
// One thread per (m_tile, k_block, lane) chunk of 16 bf16 (32 bytes).
// A 16x32 bf16 fragment order used here (contiguous k, same as B): lane L holds row M = L%16;
//   element j (j=0..15): K = (L/16)*16 + j
// ---------------------------------------------------------------------------
__global__ __launch_bounds__(256) void swizzleA_kernel(
    const float* __restrict__ x, unsigned short* __restrict__ Asw_u16)
{
    __bf16* __restrict__ Asw = (__bf16*)Asw_u16;
    size_t tid = (size_t)blockIdx.x * blockDim.x + threadIdx.x;
    int    lane  = (int)(tid & 31);
    size_t tmp   = tid >> 5;
    int    kb    = (int)(tmp & (KB - 1));
    int    mtile = (int)(tmp >> 7);          // KB == 128 -> 7 bits
    int    m     = mtile * 16 + (lane & 15);
    int    half  = lane >> 4;

    const float* src = x + (size_t)m * DIN + kb * 32;
    __bf16*      dst = Asw + tid * 16;
#pragma unroll
    for (int j = 0; j < 16; ++j) {
        int k = half * 16 + j;
        dst[j] = (__bf16)src[k];
    }
}

// ---------------------------------------------------------------------------
// Stage 1b: idx (i32, [DOUT][DIN]) + codebook -> B_sw (bf16) in WMMA
// B-fragment order. B 32x16 bf16 order used here: lane L holds column n = L%16;
// its 16 elements are k = (L/16)*16 + j, j = 0..15 (consecutive!).
// ---------------------------------------------------------------------------
__global__ __launch_bounds__(256) void swizzleB_kernel(
    const int* __restrict__ widx, const float* __restrict__ cb,
    unsigned short* __restrict__ Bsw_u16)
{
    __bf16* __restrict__ Bsw = (__bf16*)Bsw_u16;
    size_t tid = (size_t)blockIdx.x * blockDim.x + threadIdx.x;
    int    lane  = (int)(tid & 31);
    size_t tmp   = tid >> 5;
    int    kb    = (int)(tmp & (KB - 1));
    int    ntile = (int)(tmp >> 7);
    int    n     = ntile * 16 + (lane & 15);
    int    k0    = kb * 32 + (lane >> 4) * 16;

    float c1 = cb[1], c2 = cb[2];
    const int* src = widx + (size_t)n * DIN + k0;
    __bf16*    dst = Bsw + tid * 16;
#pragma unroll
    for (int j = 0; j < 16; ++j) {
        int   id = src[j];
        float v  = (id == 0) ? 0.0f : ((id == 1) ? c1 : c2);
        dst[j] = (__bf16)v;
    }
}

// ---------------------------------------------------------------------------
// Stage 2: bf16 WMMA GEMM.  Block = 128x128, k-step 32, 256 threads = 8 waves.
// Wave grid 2(m) x 4(n): each wave owns 4 m-tiles x 2 n-tiles of 16x16 acc.
// LDS double buffer: per stage 8 KB A + 8 KB B (both pre-swizzled chunks).
// ---------------------------------------------------------------------------
__global__ __launch_bounds__(256) void gemm_wmma_bf16_kernel(
    const unsigned char* __restrict__ Asw, const unsigned char* __restrict__ Bsw,
    const float* __restrict__ bias, float* __restrict__ out)
{
    __shared__ __align__(32) unsigned char sA[2][8192];
    __shared__ __align__(32) unsigned char sB[2][8192];

    const int t    = threadIdx.x;
    const int lane = t & 31;
    const int wv   = t >> 5;      // wave id 0..7, also copy-tile id
    const int wm   = wv >> 2;     // 0..1
    const int wn   = wv & 3;      // 0..3

    const int mtile0 = blockIdx.y * 8;   // base 16-row m-tile
    const int ntile0 = blockIdx.x * 8;   // base 16-col n-tile

    // this thread's 32-byte copy slot (tile wv, lane)
    const unsigned char* gA0 = Asw + ((size_t)(mtile0 + wv) * KB) * 1024 + (size_t)lane * 32;
    const unsigned char* gB0 = Bsw + ((size_t)(ntile0 + wv) * KB) * 1024 + (size_t)lane * 32;
    const unsigned ldsOff = (unsigned)wv * 1024u + (unsigned)lane * 32u;

    auto issue_copy = [&](int stage, int kb) {
        const unsigned char* ga = gA0 + (size_t)kb * 1024;
        const unsigned char* gb = gB0 + (size_t)kb * 1024;
        unsigned char* la = &sA[stage][ldsOff];
        unsigned char* lb = &sB[stage][ldsOff];
#if USE_ASYNC
        __builtin_amdgcn_global_load_async_to_lds_b128(
            (async_v4i*)ga, (async_v4i*)la, 0, 0);
        __builtin_amdgcn_global_load_async_to_lds_b128(
            (async_v4i*)(ga + 16), (async_v4i*)(la + 16), 0, 0);
        __builtin_amdgcn_global_load_async_to_lds_b128(
            (async_v4i*)gb, (async_v4i*)lb, 0, 0);
        __builtin_amdgcn_global_load_async_to_lds_b128(
            (async_v4i*)(gb + 16), (async_v4i*)(lb + 16), 0, 0);
#else
        *(int4*)(la)      = *(const int4*)(ga);
        *(int4*)(la + 16) = *(const int4*)(ga + 16);
        *(int4*)(lb)      = *(const int4*)(gb);
        *(int4*)(lb + 16) = *(const int4*)(gb + 16);
#endif
    };

    v8f acc[4][2];
#pragma unroll
    for (int i = 0; i < 4; ++i)
#pragma unroll
        for (int j = 0; j < 2; ++j)
            acc[i][j] = {};

    const unsigned aBase = (unsigned)(wm * 4) * 1024u;
    const unsigned bBase = (unsigned)(wn * 2) * 1024u;
    const unsigned fragOff = (unsigned)lane * 32u;

    issue_copy(0, 0);

    for (int kb = 0; kb < KB; ++kb) {
        const int cur = kb & 1;
#if USE_ASYNC
        __builtin_amdgcn_s_wait_asynccnt(0);
#endif
        __syncthreads();   // stage `cur` fully resident; stage cur^1 fully consumed
        if (kb + 1 < KB) issue_copy(cur ^ 1, kb + 1);

        v16bf afr[4], bfr[2];
#pragma unroll
        for (int i = 0; i < 4; ++i)
            afr[i] = *(const v16bf*)(&sA[cur][aBase + (unsigned)i * 1024u + fragOff]);
#pragma unroll
        for (int j = 0; j < 2; ++j)
            bfr[j] = *(const v16bf*)(&sB[cur][bBase + (unsigned)j * 1024u + fragOff]);

#pragma unroll
        for (int i = 0; i < 4; ++i)
#pragma unroll
            for (int j = 0; j < 2; ++j)
                acc[i][j] = __builtin_amdgcn_wmma_f32_16x16x32_bf16(
                    false, afr[i], false, bfr[j],
                    (short)0, acc[i][j], false, false);
    }

    // Epilogue: C/D 16x16 f32 layout — VGPR r: lanes 0-15 -> M=r, N=lane;
    // lanes 16-31 -> M=8+r, N=lane-16.
    const int lmod = lane & 15;
    const int lhi  = lane >> 4;
#pragma unroll
    for (int j = 0; j < 2; ++j) {
        const int col = blockIdx.x * 128 + (wn * 2 + j) * 16 + lmod;
        const float bv = bias[col];
#pragma unroll
        for (int i = 0; i < 4; ++i) {
            const int m0 = blockIdx.y * 128 + (wm * 4 + i) * 16 + lhi * 8;
            float* o = out + (size_t)m0 * DOUT + col;
#pragma unroll
            for (int r = 0; r < 8; ++r)
                o[(size_t)r * DOUT] = acc[i][j][r] + bv;
        }
    }
}

// ---------------------------------------------------------------------------
// Fallback (only if d_ws is too small): one thread per output element.
// ---------------------------------------------------------------------------
__global__ __launch_bounds__(256) void naive_kernel(
    const float* __restrict__ x, const int* __restrict__ widx,
    const float* __restrict__ cb, const float* __restrict__ bias,
    float* __restrict__ out)
{
    int n = blockIdx.x * 256 + threadIdx.x;   // 0..DOUT-1
    int m = blockIdx.y;                       // 0..MROWS-1
    if (n >= DOUT) return;
    float c1 = cb[1], c2 = cb[2];
    const float* xr = x + (size_t)m * DIN;
    const int*   wr = widx + (size_t)n * DIN;
    float s = 0.f;
    for (int k = 0; k < DIN; ++k) {
        int id = wr[k];
        float w = (id == 0) ? 0.f : ((id == 1) ? c1 : c2);
        s += xr[k] * w;
    }
    out[(size_t)m * DOUT + n] = s + bias[n];
}

extern "C" void kernel_launch(void* const* d_in, const int* in_sizes, int n_in,
                              void* d_out, int out_size, void* d_ws, size_t ws_size,
                              hipStream_t stream)
{
    const float* x    = (const float*)d_in[0];
    const int*   widx = (const int*)d_in[1];
    const float* cb   = (const float*)d_in[2];
    const float* bias = (const float*)d_in[3];
    float*       out  = (float*)d_out;

    const size_t aBytes = (size_t)MROWS * DIN * sizeof(unsigned short); // 64 MiB
    const size_t bBytes = (size_t)DOUT  * DIN * sizeof(unsigned short); // 32 MiB

    if (ws_size < aBytes + bBytes) {
        dim3 g(DOUT / 256, MROWS);
        naive_kernel<<<g, 256, 0, stream>>>(x, widx, cb, bias, out);
        return;
    }

    unsigned short* Asw = (unsigned short*)d_ws;
    unsigned short* Bsw = (unsigned short*)((char*)d_ws + aBytes);

    // A chunks: (M/16)*(K/32)*32 lanes = 2,097,152 threads
    swizzleA_kernel<<<8192, 256, 0, stream>>>(x, Asw);
    // B chunks: (N/16)*(K/32)*32 lanes = 1,048,576 threads
    swizzleB_kernel<<<4096, 256, 0, stream>>>(widx, cb, Bsw);

    dim3 grid(DOUT / 128, MROWS / 128);   // 32 x 64 blocks
    gemm_wmma_bf16_kernel<<<grid, 256, 0, stream>>>(
        (const unsigned char*)Asw, (const unsigned char*)Bsw, bias, out);
}
